// Model1_22874995818843
// MI455X (gfx1250) — compile-verified, matrix operand order checked by reading, not run on hardware
//
#include <hip/hip_runtime.h>
#include <hip/hip_bf16.h>
#include <math.h>

// ---------------------------------------------------------------------------
// Model dims (match reference)
// ---------------------------------------------------------------------------
#define BB    32
#define NN    1024
#define SS    128
#define KK    32
#define HH    8
#define DIMC  512
#define DHH   64
#define CLSN  40
#define EPSF  1e-5f
#define RAD2  0.25f

typedef __attribute__((ext_vector_type(16))) _Float16 v16h;
typedef __attribute__((ext_vector_type(8)))  _Float16 v8h;
typedef __attribute__((ext_vector_type(2)))  _Float16 h2;
typedef __attribute__((ext_vector_type(8)))  float    v8f;

// Force wave-uniform values into SGPRs so tile decode / branches scalarize.
#define RFL(x) __builtin_amdgcn_readfirstlane(x)

// ---------------------------------------------------------------------------
// WMMA fragment helpers (wave32, 16x16x32 f16 -> f32)
//
// A fragment (16x32, row-major, stride ld):
//   lane<16 : row=lane,    halves = A[row][0..7], A[row][16..23]
//   lane>=16: row=lane-16, halves = A[row][8..15], A[row][24..31]
// B fragment is loaded from a TRANSPOSED [N][K] row-major buffer with the
// same per-lane pattern (lane selects output column N).
// D fragment (16x16 f32, 8 regs): lane<16 col=lane rows 0..7;
// lane>=16 col=lane-16 rows 8..15.
// ---------------------------------------------------------------------------
__device__ __forceinline__ v16h load_frag(const _Float16* base, int ld) {
  int lane = threadIdx.x & 31;
  int row  = lane & 15;
  int koff = (lane >> 4) << 3;           // 0 or 8
  const _Float16* p = base + (size_t)row * ld + koff;
  v8h lo = *reinterpret_cast<const v8h*>(p);
  v8h hi = *reinterpret_cast<const v8h*>(p + 16);
  return __builtin_shufflevector(lo, hi, 0,1,2,3,4,5,6,7,8,9,10,11,12,13,14,15);
}

// C[16x16] += A[16xK] * Bt[16(N) x K]  (both operands row-major w/ given ld)
__device__ __forceinline__ v8f wmma_accum(const _Float16* A, int lda,
                                          const _Float16* Bt, int ldb,
                                          int K, v8f acc) {
  for (int k = 0; k < K; k += 32) {
    v16h a = load_frag(A + k, lda);
    v16h b = load_frag(Bt + k, ldb);
    acc = __builtin_amdgcn_wmma_f32_16x16x32_f16(
        false, a, false, b, (short)0, acc, false, false);
  }
  return acc;
}

// Async global->LDS copy of one 16-byte chunk (ASYNCcnt-tracked DMA path).
// ldsoff is the byte offset within the workgroup's dynamic LDS (no static LDS
// is used in callers, so dynamic LDS starts at offset 0).
__device__ __forceinline__ void async_copy16(unsigned ldsoff, const void* gsrc) {
  unsigned long long ga = (unsigned long long)gsrc;
  asm volatile("global_load_async_to_lds_b128 %0, %1, off"
               :: "v"(ldsoff), "v"(ga) : "memory");
}
__device__ __forceinline__ void async_wait0() {
  asm volatile("s_wait_asynccnt 0x0" ::: "memory");
}

// ---------------------------------------------------------------------------
// Small utility kernels
// ---------------------------------------------------------------------------
__global__ void cvt_kernel(const float* __restrict__ src,
                           _Float16* __restrict__ dst, size_t n) {
  size_t t = (size_t)blockIdx.x * blockDim.x + threadIdx.x;
  if (t < n) dst[t] = (_Float16)src[t];
}

// pts[(b*N+n)*512+c] = xyz . emb_w[c] + emb_b[c];  x is [B,3,N]
__global__ void embed_kernel(const float* __restrict__ x,
                             const float* __restrict__ w,
                             const float* __restrict__ bias,
                             float* __restrict__ pts) {
  size_t t = (size_t)blockIdx.x * blockDim.x + threadIdx.x;
  if (t >= (size_t)BB * NN * DIMC) return;
  int c = (int)(t & (DIMC - 1));
  size_t bn = t >> 9;
  int n = (int)(bn & (NN - 1));
  int b = (int)(bn >> 10);
  const float* xb = x + (size_t)b * 3 * NN;
  pts[t] = bias[c] + xb[n] * w[c*3+0] + xb[NN+n] * w[c*3+1] + xb[2*NN+n] * w[c*3+2];
}

// Farthest point sampling: one workgroup per batch, deterministic start at 0.
__global__ void fps_kernel(const float* __restrict__ x, int* __restrict__ fpsidx) {
  __shared__ float dist[NN];
  __shared__ float rv[256];
  __shared__ int   ri[256];
  __shared__ int   curfar;
  int b = blockIdx.x;
  const float* xb = x + (size_t)b * 3 * NN;
  for (int t = threadIdx.x; t < NN; t += 256) dist[t] = 1e10f;
  if (threadIdx.x == 0) curfar = 0;
  __syncthreads();
  for (int s = 0; s < SS; ++s) {
    int far = curfar;
    if (threadIdx.x == 0) fpsidx[b * SS + s] = far;
    float cx = xb[far], cy = xb[NN + far], cz = xb[2*NN + far];
    float bestv = -1.0f; int besti = 0;
    for (int t = threadIdx.x; t < NN; t += 256) {
      float dx = xb[t] - cx, dy = xb[NN+t] - cy, dz = xb[2*NN+t] - cz;
      float nd = fminf(dist[t], dx*dx + dy*dy + dz*dz);
      dist[t] = nd;
      if (nd > bestv) { bestv = nd; besti = t; }
    }
    rv[threadIdx.x] = bestv; ri[threadIdx.x] = besti;
    __syncthreads();
    for (int st = 128; st > 0; st >>= 1) {
      if (threadIdx.x < st) {
        bool take = (rv[threadIdx.x+st] > rv[threadIdx.x]) ||
                    (rv[threadIdx.x+st] == rv[threadIdx.x] &&
                     ri[threadIdx.x+st] < ri[threadIdx.x]);
        if (take) { rv[threadIdx.x] = rv[threadIdx.x+st]; ri[threadIdx.x] = ri[threadIdx.x+st]; }
      }
      __syncthreads();
    }
    if (threadIdx.x == 0) curfar = ri[0];
    __syncthreads();
  }
}

// Ball query (first K indices in ascending order within radius) + gather:
// xl[(inst*512+c)*32+k] = pts[(b*N + sel[k])*512 + c]
__global__ void group_kernel(const float* __restrict__ x,
                             const float* __restrict__ pts,
                             const int* __restrict__ fpsidx,
                             float* __restrict__ xl) {
  __shared__ float d2[NN];
  __shared__ int sel[KK];
  int b = blockIdx.x / SS, s = blockIdx.x % SS;
  const float* xb = x + (size_t)b * 3 * NN;
  int ci = fpsidx[b * SS + s];
  float cx = xb[ci], cy = xb[NN + ci], cz = xb[2*NN + ci];
  for (int t = threadIdx.x; t < NN; t += 256) {
    float dx = xb[t] - cx, dy = xb[NN+t] - cy, dz = xb[2*NN+t] - cz;
    d2[t] = dx*dx + dy*dy + dz*dz;
  }
  __syncthreads();
  if (threadIdx.x == 0) {
    int cnt = 0, first = 0; bool have = false;
    for (int t = 0; t < NN && cnt < KK; ++t) {
      if (d2[t] <= RAD2) {
        if (!have) { first = t; have = true; }
        sel[cnt++] = t;
      }
    }
    for (; cnt < KK; ++cnt) sel[cnt] = first;
  }
  __syncthreads();
  size_t inst = (size_t)b * SS + s;
  for (int t = threadIdx.x; t < DIMC * KK; t += 256) {
    int c = t >> 5, k = t & 31;
    xl[(inst * DIMC + c) * KK + k] = pts[((size_t)b * NN + sel[k]) * DIMC + c];
  }
}

// Per-channel batchnorm stats over [inst, c, n]; one workgroup per channel.
__global__ void bn_stats_kernel(const float* __restrict__ x, int inst, int n,
                                float* __restrict__ mean, float* __restrict__ invstd) {
  __shared__ float s1[256], s2[256];
  int c = blockIdx.x;
  size_t total = (size_t)inst * n;
  float a = 0.f, bsum = 0.f;
  for (size_t t = threadIdx.x; t < total; t += 256) {
    size_t it = t / n, j = t % n;
    float v = x[(it * DIMC + c) * (size_t)n + j];
    a += v; bsum += v * v;
  }
  s1[threadIdx.x] = a; s2[threadIdx.x] = bsum;
  __syncthreads();
  for (int st = 128; st > 0; st >>= 1) {
    if (threadIdx.x < st) { s1[threadIdx.x] += s1[threadIdx.x+st]; s2[threadIdx.x] += s2[threadIdx.x+st]; }
    __syncthreads();
  }
  if (threadIdx.x == 0) {
    float m = s1[0] / (float)total;
    float v = s2[0] / (float)total - m * m;
    mean[c] = m; invstd[c] = rsqrtf(v + EPSF);
  }
}

// ---------------------------------------------------------------------------
// BN + QKV grouped GEMM (WMMA). One workgroup (8 waves) per instance.
// xnT (dyn LDS): normalized activations transposed [NC][512] f16 so they are
// the row-major [N][K] "Bt" operand. Outputs:
//   TA[inst][h][j][d] from wk-output (reference swaps: q = split(k_))
//   TB[inst][h][j][d] from wq-output
//   Vp[inst][c][j]    from wv-output (plain layout, used as strided Bt later)
// ---------------------------------------------------------------------------
template <int NC>
__global__ void qkv_kernel(const float* __restrict__ x,
                           const float* __restrict__ mean,
                           const float* __restrict__ invstd,
                           const float* __restrict__ gamma,
                           const float* __restrict__ beta,
                           const _Float16* __restrict__ wk,
                           const _Float16* __restrict__ wq,
                           const _Float16* __restrict__ wv,
                           _Float16* __restrict__ TA,
                           _Float16* __restrict__ TB,
                           _Float16* __restrict__ Vp) {
  extern __shared__ char smem[];
  _Float16* xnT = (_Float16*)smem;                 // [NC][512]
  size_t inst = blockIdx.x;
  for (int t = threadIdx.x; t < DIMC * NC; t += 256) {
    int c = t / NC, j = t % NC;
    float v = x[(inst * DIMC + c) * NC + j];
    xnT[(size_t)j * DIMC + c] =
        (_Float16)(gamma[c] * (v - mean[c]) * invstd[c] + beta[c]);
  }
  __syncthreads();
  const int NT = NC / 16;
  int wave = threadIdx.x >> 5, lane = threadIdx.x & 31;
  int col = lane & 15, r0 = (lane >> 4) << 3;
  int tiles = 3 * HH * 4 * NT;
  for (int t = wave; t < tiles; t += 8) {
    int ts  = RFL(t);                 // wave-uniform -> SGPR
    int m   = ts / (HH * 4 * NT);
    int rm  = ts % (HH * 4 * NT);
    int g   = rm / (4 * NT);
    int rm2 = rm % (4 * NT);
    int mt  = rm2 / NT, nt = rm2 % NT;
    const _Float16* W = (m == 0 ? wk : (m == 1 ? wq : wv)) +
                        ((size_t)g * DHH + mt * 16) * DHH;
    const _Float16* Bt = xnT + (size_t)(nt * 16) * DIMC + g * DHH;
    v8f acc = {};
    acc = wmma_accum(W, DHH, Bt, DIMC, DHH, acc);
    int jj = nt * 16 + col;
    if (m == 2) {                    // scalar branch (m is uniform)
#pragma unroll
      for (int r = 0; r < 8; ++r) {
        int c = g * DHH + mt * 16 + r0 + r;
        Vp[(inst * DIMC + c) * NC + jj] = (_Float16)acc[r];
      }
    } else {
      _Float16* T = (m == 0) ? TA : TB;
#pragma unroll
      for (int r = 0; r < 8; ++r) {
        int c = g * DHH + mt * 16 + r0 + r;
        int h = c & 7, d = c >> 3;   // channel split c = d*H + h
        T[(((inst * HH) + h) * (size_t)NC + jj) * DHH + d] = (_Float16)acc[r];
      }
    }
  }
}

// ---------------------------------------------------------------------------
// Attention: one workgroup per (inst, head). S = Q Kt / 8 (WMMA), softmax in
// LDS, O = P V (WMMA). Writes OT[inst][i][h*64+d] f16 (transposed for proj).
// ---------------------------------------------------------------------------
template <int NC>
__global__ void attn_kernel(const _Float16* __restrict__ TA,
                            const _Float16* __restrict__ TB,
                            const _Float16* __restrict__ Vp,
                            _Float16* __restrict__ OT) {
  extern __shared__ char smem[];
  float*    Sm = (float*)smem;                       // [NC][NC]
  _Float16* Pm = (_Float16*)(smem + (size_t)NC * NC * 4); // [NC][NC]
  size_t inst = blockIdx.x >> 3;
  int head = blockIdx.x & 7;
  int wave = threadIdx.x >> 5, lane = threadIdx.x & 31;
  int col = lane & 15, r0 = (lane >> 4) << 3;
  const int NT = NC / 16;
  const _Float16* Abase = TA + ((inst * HH) + head) * (size_t)NC * DHH;
  const _Float16* Bbase = TB + ((inst * HH) + head) * (size_t)NC * DHH;
  for (int t = wave; t < NT * NT; t += 8) {
    int ts = RFL(t);
    int mt = ts / NT, nt = ts % NT;
    v8f acc = {};
    acc = wmma_accum(Abase + (size_t)mt * 16 * DHH, DHH,
                     Bbase + (size_t)nt * 16 * DHH, DHH, DHH, acc);
    int jj = nt * 16 + col;
#pragma unroll
    for (int r = 0; r < 8; ++r)
      Sm[(mt * 16 + r0 + r) * NC + jj] = acc[r] * 0.125f;
  }
  __syncthreads();
  for (int r = threadIdx.x; r < NC; r += 256) {
    float mx = -1e30f;
    for (int j = 0; j < NC; ++j) mx = fmaxf(mx, Sm[r * NC + j]);
    float ssum = 0.f;
    for (int j = 0; j < NC; ++j) {
      float e = __expf(Sm[r * NC + j] - mx);
      Sm[r * NC + j] = e; ssum += e;
    }
    float inv = 1.f / ssum;
    for (int j = 0; j < NC; ++j) Pm[r * NC + j] = (_Float16)(Sm[r * NC + j] * inv);
  }
  __syncthreads();
  // O[i][d] = sum_j P[i][j] * v'[j][d];  Bt[d][j] = Vp[(inst*512 + d*8+head)][j]
  const _Float16* Vbase = Vp + (inst * DIMC + head) * (size_t)NC;
  for (int t = wave; t < NT * 4; t += 8) {
    int ts = RFL(t);
    int mt = ts / 4, nt = ts % 4;
    v8f acc = {};
    acc = wmma_accum(Pm + (size_t)mt * 16 * NC, NC,
                     Vbase + (size_t)nt * 16 * (HH * NC), HH * NC, NC, acc);
    int d = nt * 16 + col;
#pragma unroll
    for (int r = 0; r < 8; ++r) {
      int i = mt * 16 + r0 + r;
      OT[(inst * NC + i) * (size_t)DIMC + head * DHH + d] = (_Float16)acc[r];
    }
  }
}

// ---------------------------------------------------------------------------
// Output projection: att = Wo * O + bo + x.  The B panel (OT slice, NC*512
// f16) is reused by all 32 M-tiles, so it is staged into LDS once via the
// ASYNCcnt async global->LDS DMA path, then all WMMAs read it from LDS.
// ---------------------------------------------------------------------------
template <int NC>
__global__ void proj_kernel(const _Float16* __restrict__ Wo,
                            const float* __restrict__ bo,
                            const _Float16* __restrict__ OT,
                            const float* __restrict__ xin,
                            float* __restrict__ attout) {
  extern __shared__ char smem[];
  _Float16* BtL = (_Float16*)smem;                 // [NC][512] f16
  size_t inst = blockIdx.x;
  const _Float16* src = OT + inst * (size_t)NC * DIMC;
  const int chunks = NC * 64;                      // 16B chunks in the panel
  for (int u = threadIdx.x; u < chunks; u += 256) {
    async_copy16((unsigned)(u * 16), (const char*)src + (size_t)u * 16);
  }
  async_wait0();
  __syncthreads();

  int wave = threadIdx.x >> 5, lane = threadIdx.x & 31;
  int col = lane & 15, r0 = (lane >> 4) << 3;
  const int NT = NC / 16;
  for (int t = wave; t < 32 * NT; t += 8) {
    int ts = RFL(t);
    int mt = ts / NT, nt = ts % NT;
    const _Float16* A = Wo + (size_t)mt * 16 * DIMC;
    __builtin_prefetch(A, 0, 3);
    v8f acc = {};
    acc = wmma_accum(A, DIMC, BtL + (size_t)nt * 16 * DIMC, DIMC, DIMC, acc);
    int jj = nt * 16 + col;
#pragma unroll
    for (int r = 0; r < 8; ++r) {
      int c = mt * 16 + r0 + r;
      size_t o = (inst * DIMC + c) * NC + jj;
      attout[o] = acc[r] + bo[c] + xin[o];
    }
  }
}

// ---------------------------------------------------------------------------
// FFN: xout = att + W2 * relu(gconv(bn(att), W1))    (all GEMMs via WMMA)
// ---------------------------------------------------------------------------
template <int NC>
__global__ void ffn_kernel(const float* __restrict__ att,
                           const float* __restrict__ mean,
                           const float* __restrict__ invstd,
                           const float* __restrict__ gamma,
                           const float* __restrict__ beta,
                           const _Float16* __restrict__ W1,
                           const _Float16* __restrict__ W2,
                           float* __restrict__ xout) {
  extern __shared__ char smem[];
  _Float16* anT = (_Float16*)smem;                     // [NC][512]
  _Float16* fT  = anT + (size_t)NC * DIMC;             // [NC][512]
  size_t inst = blockIdx.x;
  for (int t = threadIdx.x; t < DIMC * NC; t += 256) {
    int c = t / NC, j = t % NC;
    float v = att[(inst * DIMC + c) * NC + j];
    anT[(size_t)j * DIMC + c] =
        (_Float16)(gamma[c] * (v - mean[c]) * invstd[c] + beta[c]);
  }
  __syncthreads();
  const int NT = NC / 16;
  int wave = threadIdx.x >> 5, lane = threadIdx.x & 31;
  int col = lane & 15, r0 = (lane >> 4) << 3;
  for (int t = wave; t < HH * 4 * NT; t += 8) {
    int ts = RFL(t);
    int g = ts / (4 * NT);
    int rm = ts % (4 * NT);
    int mt = rm / NT, nt = rm % NT;
    v8f acc = {};
    acc = wmma_accum(W1 + ((size_t)g * DHH + mt * 16) * DHH, DHH,
                     anT + (size_t)(nt * 16) * DIMC + g * DHH, DIMC, DHH, acc);
    int jj = nt * 16 + col;
    // consecutive r -> adjacent channels: pack two f16 per 32-bit DS store
    int cbase = g * DHH + mt * 16 + r0;
#pragma unroll
    for (int r = 0; r < 8; r += 2) {
      h2 pk;
      pk.x = (_Float16)fmaxf(acc[r], 0.f);
      pk.y = (_Float16)fmaxf(acc[r + 1], 0.f);
      *reinterpret_cast<h2*>(&fT[(size_t)jj * DIMC + cbase + r]) = pk;
    }
  }
  __syncthreads();
  for (int t = wave; t < 32 * NT; t += 8) {
    int ts = RFL(t);
    int mt = ts / NT, nt = ts % NT;
    const _Float16* A = W2 + (size_t)mt * 16 * DIMC;
    __builtin_prefetch(A, 0, 3);
    v8f acc = {};
    acc = wmma_accum(A, DIMC, fT + (size_t)(nt * 16) * DIMC, DIMC, DIMC, acc);
    int jj = nt * 16 + col;
#pragma unroll
    for (int r = 0; r < 8; ++r) {
      int c = mt * 16 + r0 + r;
      size_t o = (inst * DIMC + c) * NC + jj;
      xout[o] = att[o] + acc[r];
    }
  }
}

// ---------------------------------------------------------------------------
// Pooling + head
// ---------------------------------------------------------------------------
__global__ void maxk_kernel(const float* __restrict__ xl, float* __restrict__ xg) {
  size_t t = (size_t)blockIdx.x * blockDim.x + threadIdx.x;
  if (t >= (size_t)BB * DIMC * SS) return;
  int s = (int)(t % SS);
  size_t bc = t / SS;
  int c = (int)(bc % DIMC), b = (int)(bc / DIMC);
  const float* p = xl + (((size_t)b * SS + s) * DIMC + c) * KK;
  float m = p[0];
  for (int k = 1; k < KK; ++k) m = fmaxf(m, p[k]);
  xg[t] = m;  // xg[b][c][s]
}

__global__ void maxs_kernel(const float* __restrict__ xg, float* __restrict__ feat) {
  size_t t = (size_t)blockIdx.x * blockDim.x + threadIdx.x;
  if (t >= (size_t)BB * DIMC) return;
  const float* p = xg + t * SS;
  float m = p[0];
  for (int s = 1; s < SS; ++s) m = fmaxf(m, p[s]);
  feat[t] = m;  // feat[b][c]
}

__global__ void fc_kernel(const float* __restrict__ in, const float* __restrict__ w,
                          const float* __restrict__ bias, int INF, int OUTF,
                          float* __restrict__ out) {
  int t = blockIdx.x * blockDim.x + threadIdx.x;
  if (t >= BB * OUTF) return;
  int o = t % OUTF, b = t / OUTF;
  const float* ip = in + (size_t)b * INF;
  const float* wp = w + (size_t)o * INF;
  float a = bias[o];
  for (int i = 0; i < INF; ++i) a += ip[i] * wp[i];
  out[t] = a;
}

// BN over batch (32 samples = 1 wave) + exact GELU, in place.
__global__ void bn_gelu_kernel(float* __restrict__ h, const float* __restrict__ g,
                               const float* __restrict__ beta, int OUTF) {
  int o = blockIdx.x;
  int b = threadIdx.x;  // 32 threads
  float v = h[(size_t)b * OUTF + o];
  float s = v, s2 = v * v;
  for (int m = 16; m > 0; m >>= 1) { s += __shfl_xor(s, m); s2 += __shfl_xor(s2, m); }
  float mean = s * (1.f / 32.f);
  float var  = s2 * (1.f / 32.f) - mean * mean;
  float y = g[o] * (v - mean) * rsqrtf(var + EPSF) + beta[o];
  h[(size_t)b * OUTF + o] = 0.5f * y * (1.f + erff(y * 0.70710678118654752f));
}

// ---------------------------------------------------------------------------
// Host orchestration
// ---------------------------------------------------------------------------
extern "C" void kernel_launch(void* const* d_in, const int* in_sizes, int n_in,
                              void* d_out, int out_size, void* d_ws, size_t ws_size,
                              hipStream_t stream) {
  // setup_inputs() dict order
  const float* x      = (const float*)d_in[0];
  const float* emb_w  = (const float*)d_in[1];
  const float* emb_b  = (const float*)d_in[2];
  const float* l_ng = (const float*)d_in[3],  *l_fg = (const float*)d_in[4];
  const float* l_nb = (const float*)d_in[5],  *l_fb = (const float*)d_in[6];
  const float* l_wk = (const float*)d_in[7],  *l_wq = (const float*)d_in[8];
  const float* l_wv = (const float*)d_in[9],  *l_w1 = (const float*)d_in[10];
  const float* l_wo = (const float*)d_in[11], *l_w2 = (const float*)d_in[12];
  const float* l_bo = (const float*)d_in[13];
  const float* g_ng = (const float*)d_in[14], *g_fg = (const float*)d_in[15];
  const float* g_nb = (const float*)d_in[16], *g_fb = (const float*)d_in[17];
  const float* g_wk = (const float*)d_in[18], *g_wq = (const float*)d_in[19];
  const float* g_wv = (const float*)d_in[20], *g_w1 = (const float*)d_in[21];
  const float* g_wo = (const float*)d_in[22], *g_w2 = (const float*)d_in[23];
  const float* g_bo = (const float*)d_in[24];
  const float* c1_w = (const float*)d_in[25], *c1_b = (const float*)d_in[26];
  const float* c1_g = (const float*)d_in[27], *c1_be = (const float*)d_in[28];
  const float* c2_w = (const float*)d_in[29], *c2_b = (const float*)d_in[30];
  const float* c2_g = (const float*)d_in[31], *c2_be = (const float*)d_in[32];
  const float* c3_w = (const float*)d_in[33], *c3_b = (const float*)d_in[34];
  float* out = (float*)d_out;

  const size_t LINST = (size_t)BB * SS;     // 4096
  const size_t GINST = BB;                  // 32
  const size_t WSMALL = 3ull * HH * DHH * DHH;     // 98304 (wk/wq/wv/w1)
  const size_t WBIG   = 3ull * DIMC * DIMC;        // 786432 (wo/w2)

  char* basep = (char*)d_ws;
  size_t off = 0;
  auto carve = [&](size_t bytes) -> void* {
    void* p = basep + off;
    off = (off + bytes + 255) & ~(size_t)255;
    return p;
  };
  _Float16* wk_l = (_Float16*)carve(WSMALL * 2);
  _Float16* wq_l = (_Float16*)carve(WSMALL * 2);
  _Float16* wv_l = (_Float16*)carve(WSMALL * 2);
  _Float16* w1_l = (_Float16*)carve(WSMALL * 2);
  _Float16* wo_l = (_Float16*)carve(WBIG * 2);
  _Float16* w2_l = (_Float16*)carve(WBIG * 2);
  _Float16* wk_g = (_Float16*)carve(WSMALL * 2);
  _Float16* wq_g = (_Float16*)carve(WSMALL * 2);
  _Float16* wv_g = (_Float16*)carve(WSMALL * 2);
  _Float16* w1_g = (_Float16*)carve(WSMALL * 2);
  _Float16* wo_g = (_Float16*)carve(WBIG * 2);
  _Float16* w2_g = (_Float16*)carve(WBIG * 2);
  float* pts    = (float*)carve((size_t)BB * NN * DIMC * 4);
  int*   fpsidx = (int*)carve((size_t)BB * SS * 4);
  float* xl     = (float*)carve(LINST * DIMC * KK * 4);
  float* attl   = (float*)carve(LINST * DIMC * KK * 4);
  _Float16* TA  = (_Float16*)carve(LINST * HH * KK * DHH * 2);
  _Float16* TB  = (_Float16*)carve(LINST * HH * KK * DHH * 2);
  _Float16* Vp  = (_Float16*)carve(LINST * DIMC * KK * 2);
  _Float16* OT  = (_Float16*)carve(LINST * KK * DIMC * 2);
  float* xg     = (float*)carve(GINST * DIMC * SS * 4);
  float* attg   = (float*)carve(GINST * DIMC * SS * 4);
  float* meanA  = (float*)carve(DIMC * 4);
  float* invA   = (float*)carve(DIMC * 4);
  float* meanB  = (float*)carve(DIMC * 4);
  float* invB   = (float*)carve(DIMC * 4);
  float* feat   = (float*)carve((size_t)BB * DIMC * 4);
  float* h1     = (float*)carve((size_t)BB * 512 * 4);
  float* h2v    = (float*)carve((size_t)BB * 256 * 4);

  auto cvt = [&](const float* s, _Float16* d, size_t n) {
    cvt_kernel<<<(unsigned)((n + 255) / 256), 256, 0, stream>>>(s, d, n);
  };
  cvt(l_wk, wk_l, WSMALL); cvt(l_wq, wq_l, WSMALL); cvt(l_wv, wv_l, WSMALL);
  cvt(l_w1, w1_l, WSMALL); cvt(l_wo, wo_l, WBIG);   cvt(l_w2, w2_l, WBIG);
  cvt(g_wk, wk_g, WSMALL); cvt(g_wq, wq_g, WSMALL); cvt(g_wv, wv_g, WSMALL);
  cvt(g_w1, w1_g, WSMALL); cvt(g_wo, wo_g, WBIG);   cvt(g_w2, w2_g, WBIG);

  embed_kernel<<<(unsigned)(((size_t)BB*NN*DIMC + 255)/256), 256, 0, stream>>>(x, emb_w, emb_b, pts);
  fps_kernel<<<BB, 256, 0, stream>>>(x, fpsidx);
  group_kernel<<<(unsigned)LINST, 256, 0, stream>>>(x, pts, fpsidx, xl);

  // ---- Local stage: 3 blocks on [4096, 512, 32] ----
  for (int blk = 0; blk < 3; ++blk) {
    size_t ws_off = (size_t)blk * HH * DHH * DHH;
    size_t wb_off = (size_t)blk * DIMC * DIMC;
    bn_stats_kernel<<<DIMC, 256, 0, stream>>>(xl, (int)LINST, KK, meanA, invA);
    qkv_kernel<KK><<<(unsigned)LINST, 256, KK * DIMC * 2, stream>>>(
        xl, meanA, invA, l_ng + blk * DIMC, l_nb + blk * DIMC,
        wk_l + ws_off, wq_l + ws_off, wv_l + ws_off, TA, TB, Vp);
    attn_kernel<KK><<<(unsigned)(LINST * HH), 256, KK * KK * 6, stream>>>(TA, TB, Vp, OT);
    proj_kernel<KK><<<(unsigned)LINST, 256, KK * DIMC * 2, stream>>>(
        wo_l + wb_off, l_bo + blk * DIMC, OT, xl, attl);
    bn_stats_kernel<<<DIMC, 256, 0, stream>>>(attl, (int)LINST, KK, meanB, invB);
    ffn_kernel<KK><<<(unsigned)LINST, 256, 2 * KK * DIMC * 2, stream>>>(
        attl, meanB, invB, l_fg + blk * DIMC, l_fb + blk * DIMC,
        w1_l + ws_off, w2_l + wb_off, xl);
  }

  maxk_kernel<<<(unsigned)(((size_t)BB*DIMC*SS + 255)/256), 256, 0, stream>>>(xl, xg);

  // ---- Global stage: 3 blocks on [32, 512, 128] ----
  for (int blk = 0; blk < 3; ++blk) {
    size_t ws_off = (size_t)blk * HH * DHH * DHH;
    size_t wb_off = (size_t)blk * DIMC * DIMC;
    bn_stats_kernel<<<DIMC, 256, 0, stream>>>(xg, (int)GINST, SS, meanA, invA);
    qkv_kernel<SS><<<(unsigned)GINST, 256, SS * DIMC * 2, stream>>>(
        xg, meanA, invA, g_ng + blk * DIMC, g_nb + blk * DIMC,
        wk_g + ws_off, wq_g + ws_off, wv_g + ws_off, TA, TB, Vp);
    attn_kernel<SS><<<(unsigned)(GINST * HH), 256, SS * SS * 6, stream>>>(TA, TB, Vp, OT);
    proj_kernel<SS><<<(unsigned)GINST, 256, SS * DIMC * 2, stream>>>(
        wo_g + wb_off, g_bo + blk * DIMC, OT, xg, attg);
    bn_stats_kernel<<<DIMC, 256, 0, stream>>>(attg, (int)GINST, SS, meanB, invB);
    ffn_kernel<SS><<<(unsigned)GINST, 256, 2 * SS * DIMC * 2, stream>>>(
        attg, meanB, invB, g_fg + blk * DIMC, g_fb + blk * DIMC,
        w1_g + ws_off, w2_g + wb_off, xg);
  }

  maxs_kernel<<<(unsigned)(((size_t)BB*DIMC + 255)/256), 256, 0, stream>>>(xg, feat);

  // ---- Head ----
  fc_kernel<<<(BB*512 + 255)/256, 256, 0, stream>>>(feat, c1_w, c1_b, DIMC, 512, h1);
  bn_gelu_kernel<<<512, 32, 0, stream>>>(h1, c1_g, c1_be, 512);
  fc_kernel<<<(BB*256 + 255)/256, 256, 0, stream>>>(h1, c2_w, c2_b, 512, 256, h2v);
  bn_gelu_kernel<<<256, 32, 0, stream>>>(h2v, c2_g, c2_be, 256);
  fc_kernel<<<(BB*CLSN + 255)/256, 256, 0, stream>>>(h2v, c3_w, c3_b, 256, CLSN, out);
}
